// MoELayer_12043088298374
// MI455X (gfx1250) — matrix operand order repaired and checked by reading, not compiled, not run on hardware
//
#include <hip/hip_runtime.h>
#include <math.h>

typedef float    v8f   __attribute__((ext_vector_type(8)));
typedef __bf16   v16bf __attribute__((ext_vector_type(16)));
typedef unsigned u32x4 __attribute__((ext_vector_type(4)));
typedef float    f32x4 __attribute__((ext_vector_type(4)));

#define E_ 8
#define K_ 2
#define D_ 1024
#define HID_ 4096

// ---------------- bf16 split helpers (round-to-nearest-even) ----------------
__device__ __forceinline__ unsigned bf16h(float f) {
  unsigned u = __float_as_uint(f);
  return (u + 0x7FFFu + ((u >> 16) & 1u)) >> 16;   // low 16 bits = bf16
}
__device__ __forceinline__ float bf16tof(unsigned h) {
  return __uint_as_float(h << 16);
}
__device__ __forceinline__ void split1(float x, unsigned& h, unsigned& l) {
  h = bf16h(x);
  l = bf16h(x - bf16tof(h));
}
// pack floats (k, k+1) into one hi-dword + one lo-dword (element 2j = low 16)
__device__ __forceinline__ void splitpack2(float x0, float x1,
                                           unsigned& hd, unsigned& ld) {
  unsigned h0, l0, h1, l1;
  split1(x0, h0, l0);
  split1(x1, h1, l1);
  hd = h0 | (h1 << 16);
  ld = l0 | (l1 << 16);
}

// ---------------- Router: logits, top-2, softmax, per-token loss terms ------
__global__ void moe_router(const float* __restrict__ x,
                           const float* __restrict__ wg,
                           int T,
                           int* __restrict__ topi,      // [K,T]
                           float* __restrict__ pgate,   // [K,T]
                           float* __restrict__ ptoken,  // [T,E] full softmax
                           float* __restrict__ lse2) {  // [T]
  int t = blockIdx.x;
  int tid = threadIdx.x;
  int wave = tid >> 5, lane = tid & 31;
  __shared__ float ls[E_];
  const float* xr = x + (size_t)t * D_;
  float sum = 0.f;
  for (int d = lane; d < D_; d += 32)
    sum += xr[d] * wg[d * E_ + wave];
  for (int off = 16; off > 0; off >>= 1)
    sum += __shfl_xor(sum, off, 32);
  if (lane == 0) ls[wave] = sum;
  __syncthreads();
  if (tid == 0) {
    float m = ls[0];
    for (int e = 1; e < E_; ++e) m = fmaxf(m, ls[e]);
    float s = 0.f, ef[E_];
    for (int e = 0; e < E_; ++e) { ef[e] = expf(ls[e] - m); s += ef[e]; }
    float inv = 1.f / s;
    for (int e = 0; e < E_; ++e) ptoken[(size_t)t * E_ + e] = ef[e] * inv;
    float lse = m + logf(s);
    lse2[t] = lse * lse;
    int i0 = 0;
    for (int e = 1; e < E_; ++e) if (ls[e] > ls[i0]) i0 = e;
    int i1 = (i0 == 0) ? 1 : 0;
    for (int e = 0; e < E_; ++e) if (e != i0 && ls[e] > ls[i1]) i1 = e;
    float v0 = ls[i0], v1 = ls[i1];
    float e1 = expf(v1 - v0);
    float ss = 1.f + e1;
    topi[t] = i0;        topi[T + t] = i1;
    pgate[t] = 1.f / ss; pgate[T + t] = e1 / ss;
  }
}

// ------------- Deterministic reduction: prob_per_expert sums + z-loss -------
__global__ void moe_reduce(const float* __restrict__ ptoken,
                           const float* __restrict__ lse2,
                           int T, float* __restrict__ red) {  // red[E_+1]
  __shared__ float sh[256];
  int j = blockIdx.x;
  int tid = threadIdx.x;
  float s = 0.f;
  if (j < E_) { for (int t = tid; t < T; t += 256) s += ptoken[(size_t)t * E_ + j]; }
  else        { for (int t = tid; t < T; t += 256) s += lse2[t]; }
  sh[tid] = s; __syncthreads();
  for (int off = 128; off > 0; off >>= 1) {
    if (tid < off) sh[tid] += sh[tid + off];
    __syncthreads();
  }
  if (tid == 0) red[j] = sh[0];
}

// ------- Capacity assignment: exact k-major/token-order cumsum per expert ---
__global__ void moe_rank(const int* __restrict__ topi, int T, int cap,
                         int* __restrict__ slot,
                         int* __restrict__ srct,
                         int* __restrict__ keptc) {
  int tid = threadIdx.x;
  for (int i = tid; i < E_ * cap; i += blockDim.x) srct[i] = -1;
  __syncthreads();
  if (tid < E_) {
    int e = tid, cnt = 0;
    int n = K_ * T;
    for (int f = 0; f < n; ++f) {
      if (topi[f] == e) {
        if (cnt < cap) { slot[f] = cnt; srct[e * cap + cnt] = f % T; }
        else           { slot[f] = -1; }
        cnt++;
      }
    }
    keptc[e] = cnt < cap ? cnt : cap;
  }
}

// --- Weight transpose + split: W[e][Kd][N] fp32 -> Wt{h,l}[e][N][Kd/2] dw ---
__global__ void moe_split_t(const float* __restrict__ W,
                            unsigned* __restrict__ Wth,
                            unsigned* __restrict__ Wtl,
                            int Kd, int N) {
  __shared__ float Lt[64][69];
  int e = blockIdx.z;
  int n0 = blockIdx.x * 64, k0 = blockIdx.y * 64;
  const float* We = W + (size_t)e * Kd * N;
  int tid = threadIdx.x;
#pragma unroll
  for (int l = 0; l < 4; ++l) {
    int idx = tid + l * 256;
    int r = idx >> 4, c4 = (idx & 15) * 4;
    f32x4 v = *(const f32x4*)(We + (size_t)(k0 + r) * N + n0 + c4);
    Lt[r][c4 + 0] = v.x; Lt[r][c4 + 1] = v.y;
    Lt[r][c4 + 2] = v.z; Lt[r][c4 + 3] = v.w;
  }
  __syncthreads();
  const int KdW = Kd >> 1;
  unsigned* th = Wth + (size_t)e * N * KdW;
  unsigned* tl = Wtl + (size_t)e * N * KdW;
#pragma unroll
  for (int l = 0; l < 8; ++l) {
    int idx = tid + l * 256;
    int n = idx >> 5, kb = idx & 31;
    unsigned hd, ld;
    splitpack2(Lt[2 * kb][n], Lt[2 * kb + 1][n], hd, ld);
    size_t o = (size_t)(n0 + n) * KdW + (k0 >> 1) + kb;
    th[o] = hd; tl[o] = ld;
  }
}

// ------------- Dispatch gather into split-bf16 [E, cap, D] ------------------
__global__ void moe_dispatch(const float* __restrict__ x,
                             const int* __restrict__ srct,
                             unsigned* __restrict__ Ah,
                             unsigned* __restrict__ Al) {
  int s = blockIdx.x;                 // e*cap + c
  int t = srct[s];
  int j = threadIdx.x;                // 256 threads * 4 floats = 1024
  unsigned* dh = Ah + (size_t)s * (D_ / 2);
  unsigned* dl = Al + (size_t)s * (D_ / 2);
  if (t >= 0) {
    f32x4 v = *(const f32x4*)(x + (size_t)t * D_ + 4 * j);
    unsigned h0, l0, h1, l1;
    splitpack2(v.x, v.y, h0, l0);
    splitpack2(v.z, v.w, h1, l1);
    dh[2 * j] = h0; dh[2 * j + 1] = h1;
    dl[2 * j] = l0; dl[2 * j + 1] = l1;
  } else {
    dh[2 * j] = 0u; dh[2 * j + 1] = 0u;
    dl[2 * j] = 0u; dl[2 * j + 1] = 0u;
  }
}

// ---------------- bf16x3 WMMA GEMM with async-LDS double buffering ----------
// A and B(transposed) are [rows][Kd/2] dword arrays. Block tile 128x128,
// K-chunk 32; 8 waves x 32x64 -> 24 v_wmma_f32_16x16x32_bf16 per chunk.
union Frag { v16bf v; u32x4 q[2]; };

// 2 async b128 per thread: one 128-row x 16-dword tile into LDS (stride 20 dw)
__device__ __forceinline__ void stage_async(unsigned* lptr,
                                            const unsigned* __restrict__ g,
                                            int row0, int rmax, int KdW,
                                            int kk, int tid) {
#pragma unroll
  for (int i = 0; i < 2; ++i) {
    int idx = tid + i * 256;
    int r = idx >> 2, c4 = (idx & 3) * 4;
    int grow = row0 + r;
    if (grow >= rmax) grow = rmax - 1;   // clamp; masked rows never stored
    unsigned lo = (unsigned)(size_t)(lptr + r * 20 + c4);
    unsigned long long ga =
        (unsigned long long)(size_t)(g + (size_t)grow * KdW + (kk >> 1) + c4);
    asm volatile("global_load_async_to_lds_b128 %0, %1, off"
                 :: "v"(lo), "v"(ga) : "memory");
  }
}

template <int DO_GELU, int SPLIT_OUT>
__global__ __launch_bounds__(256, 2) void moe_gemm_bf16x3(
    const unsigned* __restrict__ Ah, const unsigned* __restrict__ Al,
    const unsigned* __restrict__ Bh, const unsigned* __restrict__ Bl,
    float* __restrict__ C, unsigned short* __restrict__ Ch,
    unsigned short* __restrict__ Cl, int M, int N, int Kd) {
  __shared__ unsigned Lds[2][4][128 * 20];   // double-buffered: 80 KB
  const int e = blockIdx.z;
  const int KdW = Kd >> 1;
  const unsigned* Ahe = Ah + (size_t)e * M * KdW;
  const unsigned* Ale = Al + (size_t)e * M * KdW;
  const unsigned* Bhe = Bh + (size_t)e * N * KdW;
  const unsigned* Ble = Bl + (size_t)e * N * KdW;
  const int bm = blockIdx.y * 128, bn = blockIdx.x * 128;
  const int tid = threadIdx.x;
  const int wave = tid >> 5, lane = tid & 31;
  const int half = lane >> 4, l15 = lane & 15;
  const int m0 = (wave & 3) * 32, n0 = (wave >> 2) * 64;
  v8f acc[2][4] = {};

  const int nchunks = Kd >> 5;
  // prologue: stage chunk 0 into buffer 0 (8 async ops per wave)
  stage_async(Lds[0][0], Ahe, bm, M, KdW, 0, tid);
  stage_async(Lds[0][1], Ale, bm, M, KdW, 0, tid);
  stage_async(Lds[0][2], Bhe, bn, N, KdW, 0, tid);
  stage_async(Lds[0][3], Ble, bn, N, KdW, 0, tid);

  for (int c = 0; c < nchunks; ++c) {
    const int cur = c & 1;
    if (c + 1 < nchunks) {   // stage next chunk, then wait for current only
      const int kk = (c + 1) << 5, nxt = cur ^ 1;
      stage_async(Lds[nxt][0], Ahe, bm, M, KdW, kk, tid);
      stage_async(Lds[nxt][1], Ale, bm, M, KdW, kk, tid);
      stage_async(Lds[nxt][2], Bhe, bn, N, KdW, kk, tid);
      stage_async(Lds[nxt][3], Ble, bn, N, KdW, kk, tid);
      asm volatile("s_wait_asynccnt 0x8" ::: "memory");
    } else {
      asm volatile("s_wait_asynccnt 0x0" ::: "memory");
    }
    __syncthreads();
    const unsigned* AhL = Lds[cur][0];
    const unsigned* AlL = Lds[cur][1];
    const unsigned* BhL = Lds[cur][2];
    const unsigned* BlL = Lds[cur][3];
    // fragments: ISA 16-bit A layout (K pairs split across lane halves)
    Frag a_h[2], a_l[2];
#pragma unroll
    for (int mi = 0; mi < 2; ++mi) {
      int rb = (m0 + mi * 16 + l15) * 20;
      a_h[mi].q[0] = *(const u32x4*)&AhL[rb + half * 4];
      a_h[mi].q[1] = *(const u32x4*)&AhL[rb + 8 + half * 4];
      a_l[mi].q[0] = *(const u32x4*)&AlL[rb + half * 4];
      a_l[mi].q[1] = *(const u32x4*)&AlL[rb + 8 + half * 4];
    }
#pragma unroll
    for (int nj = 0; nj < 4; ++nj) {
      int cb = (n0 + nj * 16 + l15) * 20;
      Frag b_h, b_l;   // B: same ISA layout as A (K pairs split across lane halves)
      b_h.q[0] = *(const u32x4*)&BhL[cb + half * 4];
      b_h.q[1] = *(const u32x4*)&BhL[cb + 8 + half * 4];
      b_l.q[0] = *(const u32x4*)&BlL[cb + half * 4];
      b_l.q[1] = *(const u32x4*)&BlL[cb + 8 + half * 4];
#pragma unroll
      for (int mi = 0; mi < 2; ++mi) {
        acc[mi][nj] = __builtin_amdgcn_wmma_f32_16x16x32_bf16(
            false, a_h[mi].v, false, b_h.v, (short)0, acc[mi][nj], false, false);
        acc[mi][nj] = __builtin_amdgcn_wmma_f32_16x16x32_bf16(
            false, a_h[mi].v, false, b_l.v, (short)0, acc[mi][nj], false, false);
        acc[mi][nj] = __builtin_amdgcn_wmma_f32_16x16x32_bf16(
            false, a_l[mi].v, false, b_h.v, (short)0, acc[mi][nj], false, false);
      }
    }
    __syncthreads();   // all waves done reading buf[cur] before restaging it
  }
  // epilogue: C/D layout VGPR r -> M = r + 8*half, N = lane&15
  float* Cf = SPLIT_OUT ? nullptr : (C + (size_t)e * M * N);
  unsigned short* Chh = SPLIT_OUT ? (Ch + (size_t)e * M * N) : nullptr;
  unsigned short* Cll = SPLIT_OUT ? (Cl + (size_t)e * M * N) : nullptr;
#pragma unroll
  for (int mi = 0; mi < 2; ++mi)
#pragma unroll
    for (int nj = 0; nj < 4; ++nj)
#pragma unroll
      for (int r = 0; r < 8; ++r) {
        int grow = bm + m0 + mi * 16 + half * 8 + r;
        int gcol = bn + n0 + nj * 16 + l15;
        if (grow < M) {
          float v = acc[mi][nj][r];
          if (DO_GELU) v = 0.5f * v * (1.f + erff(v * 0.7071067811865475f));
          if (SPLIT_OUT) {
            unsigned h, l;
            split1(v, h, l);
            Chh[(size_t)grow * N + gcol] = (unsigned short)h;
            Cll[(size_t)grow * N + gcol] = (unsigned short)l;
          } else {
            Cf[(size_t)grow * N + gcol] = v;
          }
        }
      }
}

// ---------------- Combine expert outputs back to tokens ---------------------
__global__ void moe_combine(const int* __restrict__ topi,
                            const int* __restrict__ slot,
                            const float* __restrict__ pgate,
                            const float* __restrict__ O,
                            float* __restrict__ out, int T, int cap) {
  int t = blockIdx.x;
  int j = threadIdx.x;
  f32x4 o = {0.f, 0.f, 0.f, 0.f};
#pragma unroll
  for (int k = 0; k < K_; ++k) {
    int f = k * T + t;
    int c = slot[f];
    if (c >= 0) {
      int e = topi[f];
      float w = pgate[f];
      f32x4 v = *(const f32x4*)(O + (size_t)(e * cap + c) * D_ + 4 * j);
      o += w * v;
    }
  }
  *(f32x4*)(out + (size_t)t * D_ + 4 * j) = o;
}

// ---------------- Scalar losses --------------------------------------------
__global__ void moe_loss(const float* __restrict__ red,
                         const int* __restrict__ keptc, int T,
                         float* __restrict__ lb_out, float* __restrict__ z_out) {
  if (threadIdx.x == 0 && blockIdx.x == 0) {
    float invT = 1.f / (float)T;
    float lb = 0.f;
    for (int e = 0; e < E_; ++e)
      lb += (red[e] * invT) * ((float)keptc[e] / (float)(T * K_));
    *lb_out = (float)E_ * lb;
    *z_out = red[E_] * invT;
  }
}

extern "C" void kernel_launch(void* const* d_in, const int* in_sizes, int n_in,
                              void* d_out, int out_size, void* d_ws, size_t ws_size,
                              hipStream_t stream) {
  (void)n_in; (void)out_size; (void)ws_size;
  const float* x   = (const float*)d_in[0];   // [B,S,D]
  const float* wg  = (const float*)d_in[1];   // [D,E]
  const float* wfc = (const float*)d_in[2];   // [E,D,HID]
  const float* wpj = (const float*)d_in[3];   // [E,HID,D]
  float* out = (float*)d_out;

  const int T = in_sizes[0] / D_;             // 4096
  int cap = (int)((long long)5 * T / 16);     // floor(K*1.25*T/E)
  cap += cap % 2; if (cap < 2) cap = 2;       // 1280

  char* p = (char*)d_ws;
  auto carve = [&](size_t bytes) { char* r = p; p += (bytes + 255) & ~(size_t)255; return r; };
  int*      topi   = (int*)     carve((size_t)K_ * T * 4);
  float*    pgate  = (float*)   carve((size_t)K_ * T * 4);
  int*      slot   = (int*)     carve((size_t)K_ * T * 4);
  int*      srct   = (int*)     carve((size_t)E_ * cap * 4);
  int*      keptc  = (int*)     carve((size_t)E_ * 4);
  float*    ptoken = (float*)   carve((size_t)T * E_ * 4);
  float*    lse2   = (float*)   carve((size_t)T * 4);
  float*    red    = (float*)   carve((size_t)(E_ + 1) * 4);
  unsigned* AhD    = (unsigned*)carve((size_t)E_ * cap * D_ * 2);     // 20 MB
  unsigned* AlD    = (unsigned*)carve((size_t)E_ * cap * D_ * 2);
  unsigned short* Hh = (unsigned short*)carve((size_t)E_ * cap * HID_ * 2); // 80 MB
  unsigned short* Hl = (unsigned short*)carve((size_t)E_ * cap * HID_ * 2);
  float*    Ob     = (float*)   carve((size_t)E_ * cap * D_ * 4);     // 40 MB
  unsigned* Wfh    = (unsigned*)carve((size_t)E_ * D_ * HID_ * 2);    // transposed [E][HID][D/2]
  unsigned* Wfl    = (unsigned*)carve((size_t)E_ * D_ * HID_ * 2);
  unsigned* Wph    = (unsigned*)carve((size_t)E_ * HID_ * D_ * 2);    // transposed [E][D][HID/2]
  unsigned* Wpl    = (unsigned*)carve((size_t)E_ * HID_ * D_ * 2);

  // one-time weight transpose + hi/lo split
  dim3 gs1(HID_ / 64, D_ / 64, E_);
  moe_split_t<<<gs1, 256, 0, stream>>>(wfc, Wfh, Wfl, D_, HID_);
  dim3 gs2(D_ / 64, HID_ / 64, E_);
  moe_split_t<<<gs2, 256, 0, stream>>>(wpj, Wph, Wpl, HID_, D_);

  moe_router<<<T, 256, 0, stream>>>(x, wg, T, topi, pgate, ptoken, lse2);
  moe_reduce<<<E_ + 1, 256, 0, stream>>>(ptoken, lse2, T, red);
  moe_rank<<<1, 256, 0, stream>>>(topi, T, cap, slot, srct, keptc);
  moe_dispatch<<<E_ * cap, 256, 0, stream>>>(x, srct, AhD, AlD);

  dim3 g1(HID_ / 128, (cap + 127) / 128, E_);
  moe_gemm_bf16x3<1, 1><<<g1, 256, 0, stream>>>(AhD, AlD, Wfh, Wfl,
                                                nullptr, Hh, Hl, cap, HID_, D_);
  dim3 g2(D_ / 128, (cap + 127) / 128, E_);
  moe_gemm_bf16x3<0, 0><<<g2, 256, 0, stream>>>((const unsigned*)Hh, (const unsigned*)Hl,
                                                Wph, Wpl, Ob, nullptr, nullptr,
                                                cap, D_, HID_);

  moe_combine<<<T, 256, 0, stream>>>(topi, slot, pgate, Ob, out, T, cap);
  moe_loss<<<1, 1, 0, stream>>>(red, keptc, T,
                                out + (size_t)T * D_, out + (size_t)T * D_ + 1);
}
